// ptrNet_59476707115123
// MI455X (gfx1250) — matrix operand order repaired and checked by reading, not compiled, not run on hardware
//
#include <hip/hip_runtime.h>
#include <hip/hip_bf16.h>
#include <math.h>

// ---------------- Problem constants (from reference) ----------------
#define PB 128   // batch
#define PS 512   // sequence
#define PE 128   // embedding
#define PH 256   // hidden
#define P4H 1024 // 4*H
#define PKCAT 384 // E + H (concat LSTM input)
#define C_TANH 10.0f

// ---------------- CDNA5 WMMA types ----------------
typedef __attribute__((ext_vector_type(16))) __bf16 v16bf;
typedef __attribute__((ext_vector_type(8)))  __bf16 v8bf;
typedef __attribute__((ext_vector_type(8)))  float  v8f;

__device__ __forceinline__ __bf16 f2bf(float x) {
    union { float f; unsigned u; } v; v.f = x;
    unsigned r = (v.u + 0x7FFFu + ((v.u >> 16) & 1u)) >> 16;
    unsigned short s = (unsigned short)r;
    return __builtin_bit_cast(__bf16, s);
}

__device__ __forceinline__ float sigf(float x) { return 1.0f / (1.0f + __expf(-x)); }

__device__ __forceinline__ float waveRedMax(float v) {
    #pragma unroll
    for (int o = 16; o > 0; o >>= 1) v = fmaxf(v, __shfl_xor(v, o, 32));
    return v;
}
__device__ __forceinline__ float waveRedSum(float v) {
    #pragma unroll
    for (int o = 16; o > 0; o >>= 1) v += __shfl_xor(v, o, 32);
    return v;
}

__device__ __forceinline__ unsigned hash3(unsigned a, unsigned b, unsigned c) {
    unsigned h = a * 0x9E3779B1u ^ b * 0x85EBCA77u ^ c * 0xC2B2AE3Du;
    h ^= h >> 16; h *= 0x7FEB352Du; h ^= h >> 15; h *= 0x846CA68Bu; h ^= h >> 16;
    return h;
}

__device__ __forceinline__ v16bf cat8(v8bf lo, v8bf hi) {
    return __builtin_shufflevector(lo, hi, 0, 1, 2, 3, 4, 5, 6, 7,
                                           8, 9, 10, 11, 12, 13, 14, 15);
}

// ---------------- Prep kernels ----------------

// emb_in(bf16)[b][s][e]
__global__ __launch_bounds__(256) void k_embed(const float* __restrict__ inp,
                                               const float* __restrict__ emb,
                                               __bf16* __restrict__ embBf) {
    long i = (long)blockIdx.x * 256 + threadIdx.x; // over B*S*E
    if (i >= (long)PB * PS * PE) return;
    int e = (int)(i & (PE - 1));
    long bs = i >> 7;
    int s = (int)(bs & (PS - 1));
    int b = (int)(bs >> 9);
    float v = inp[(long)b * 2 * PS + s] * emb[e] +
              inp[(long)b * 2 * PS + PS + s] * emb[PE + e];
    embBf[i] = f2bf(v);
}

// WMMA B-fragment swizzle for logical [K][N] weight:
//   kc=k>>5, kr=k&31, nT=n>>4, nc=n&15
//   lane = ((kr>>4)<<4)|nc ; j = kr&15
//   idx  = ((kc*numNT + nT)*32 + lane)*16 + j     (per-lane 16 bf16 contiguous)

// Concat LSTM weight: logical Wcat[k][n] = k<E ? Wih[n][k] : Whh[n][k-E]
__global__ __launch_bounds__(256) void k_prep_cat(const float* __restrict__ Wih,
                                                  const float* __restrict__ Whh,
                                                  const float* __restrict__ bih,
                                                  const float* __restrict__ bhh,
                                                  __bf16* __restrict__ Wsw,
                                                  float* __restrict__ bsum) {
    long i = (long)blockIdx.x * 256 + threadIdx.x; // over 384*1024
    if (i < (long)PKCAT * P4H) {
        int k = (int)(i >> 10), n = (int)(i & (P4H - 1));
        float v = (k < PE) ? Wih[(long)n * PE + k] : Whh[(long)n * PH + (k - PE)];
        int kc = k >> 5, kr = k & 31;
        int nT = n >> 4, nc = n & 15;
        int lane = ((kr >> 4) << 4) | nc;
        int j = kr & 15;
        Wsw[(((long)kc * 64 + nT) * 32 + lane) * 16 + j] = f2bf(v);
    }
    if (i < P4H) bsum[i] = bih[i] + bhh[i];
}

// Square weight: logical Wt[k][n] = W[n][k]  (256x256)
__global__ __launch_bounds__(256) void k_prep_sq(const float* __restrict__ W,
                                                 __bf16* __restrict__ Wsw) {
    int i = blockIdx.x * 256 + threadIdx.x; // over 256*256
    int k = i >> 8, n = i & 255;
    float v = W[n * PH + k];
    int kc = k >> 5, kr = k & 31;
    int nT = n >> 4, nc = n & 15;
    int lane = ((kr >> 4) << 4) | nc;
    int j = kr & 15;
    Wsw[(((long)kc * 16 + nT) * 32 + lane) * 16 + j] = f2bf(v);
}

__global__ __launch_bounds__(256) void k_init(float* __restrict__ h, float* __restrict__ c,
                                              __bf16* __restrict__ hb,
                                              unsigned char* __restrict__ mask) {
    int i = blockIdx.x * 256 + threadIdx.x; // 65536 threads
    if (i < PB * PH) { h[i] = 0.f; c[i] = 0.f; hb[i] = f2bf(0.f); }
    if (i < PB * PS) mask[i] = 0;
}

__global__ __launch_bounds__(256) void k_setxin(const float* __restrict__ dec_in0,
                                                __bf16* __restrict__ xin) {
    int i = blockIdx.x * 256 + threadIdx.x; // 128*128
    xin[i] = f2bf(dec_in0[i & (PE - 1)]);
}

// ---------------- WMMA GEMM kernels ----------------

// LSTM gates: G[128,1024] = [X(128,128)|Hb(128,256)](bf16) @ Wcat(384,1024) + bsum
// Block = 8 waves; each wave: one M-tile x TWO N-tiles (shared-B via LDS).
// Grid: 32 blocks (blockIdx = N-tile-pair id). numKC = 12. LDS = 24 KB.
__global__ __launch_bounds__(256) void k_gates(const __bf16* __restrict__ X, long xRowStride,
                                               const __bf16* __restrict__ Hb,
                                               const __bf16* __restrict__ Wsw,
                                               const float* __restrict__ bias,
                                               float* __restrict__ G) {
    __shared__ __align__(32) __bf16 smem[24 * 512]; // [kc*2+w][lane][16]
    int tid  = threadIdx.x;
    int wv   = tid >> 5;
    int lane = tid & 31;
    int nT0  = blockIdx.x * 2;

    // cooperative stage of both N-tiles' B fragments (swizzled, contiguous)
    #pragma unroll
    for (int p = 0; p < 6; ++p) {               // 1536 16B-chunks total
        int c = tid + p * 256;
        int f = c >> 6;                         // fragment id 0..23 (kc*2+w)
        int kc = f >> 1, w = f & 1;
        int within = c & 63;
        const v8bf* src = (const v8bf*)(Wsw + (((long)kc * 64 + (nT0 + w)) * 512) + within * 8);
        *(v8bf*)(smem + (long)f * 512 + within * 8) = *src;
    }
    __syncthreads();

    int mTile = wv;
    int row = mTile * 16 + (lane & 15);
    int hi  = lane >> 4;
    int col0 = nT0 * 16 + (lane & 15);
    int col1 = col0 + 16;

    __builtin_prefetch(X + (long)row * xRowStride, 0, 1);
    __builtin_prefetch(Hb + row * PH, 0, 1);

    v8f acc0, acc1;
    float bv0 = bias[col0], bv1 = bias[col1];
    #pragma unroll
    for (int j = 0; j < 8; ++j) { acc0[j] = bv0; acc1[j] = bv1; }

    #pragma unroll 4
    for (int kc = 0; kc < 12; ++kc) {
        int k0 = kc * 32 + hi * 16;
        int k1 = k0 + 8;
        v8bf alo = (k0 < PE) ? *(const v8bf*)(X + (long)row * xRowStride + k0)
                             : *(const v8bf*)(Hb + row * PH + (k0 - PE));
        v8bf ahi = (k1 < PE) ? *(const v8bf*)(X + (long)row * xRowStride + k1)
                             : *(const v8bf*)(Hb + row * PH + (k1 - PE));
        v16bf a = cat8(alo, ahi);
        v16bf b0 = *(const v16bf*)(smem + ((kc * 2 + 0) * 32 + lane) * 16);
        v16bf b1 = *(const v16bf*)(smem + ((kc * 2 + 1) * 32 + lane) * 16);
        acc0 = __builtin_amdgcn_wmma_f32_16x16x32_bf16(false, a, false, b0, (short)0, acc0,
                                                       false, false);
        acc1 = __builtin_amdgcn_wmma_f32_16x16x32_bf16(false, a, false, b1, (short)0, acc1,
                                                       false, false);
    }
    #pragma unroll
    for (int j = 0; j < 8; ++j) {
        int r = mTile * 16 + j + hi * 8;
        G[(long)r * P4H + col0] = acc0[j];
        G[(long)r * P4H + col1] = acc1[j];
    }
}

// Ref projection: R[b][h][s] = sum_k Wt[k][h] * encOut[b*S+s][k] + bias[h]
// M = 65536 (4096 tiles -> 2048 pairs), N = 256 (16 tiles).
// Block = 8 waves sharing one N-tile (LDS B, 8 KB); each wave: TWO M-tiles.
// Grid: 4096 blocks; wid>>11 = nT (uniform per block), wid&2047 = mPair.
__global__ __launch_bounds__(256) void k_ref_gemm(const __bf16* __restrict__ A,
                                                  const __bf16* __restrict__ Wsw,
                                                  const float* __restrict__ bias,
                                                  float* __restrict__ R) {
    __shared__ __align__(32) __bf16 smem[8 * 512];
    int tid  = threadIdx.x;
    int lane = tid & 31;
    int wid  = blockIdx.x * 8 + (tid >> 5);
    int nT    = wid >> 11;
    int mPair = wid & 2047;

    #pragma unroll
    for (int p = 0; p < 2; ++p) {               // 512 16B-chunks total
        int c = tid + p * 256;
        int kc = c >> 6;
        int within = c & 63;
        const v8bf* src = (const v8bf*)(Wsw + (((long)kc * 16 + nT) * 512) + within * 8);
        *(v8bf*)(smem + (long)kc * 512 + within * 8) = *src;
    }
    __syncthreads();

    int r0 = mPair * 32 + (lane & 15);
    int r1 = r0 + 16;
    int col = nT * 16 + (lane & 15);
    int hi  = lane >> 4;

    __builtin_prefetch(A + (long)r0 * PH, 0, 1);
    __builtin_prefetch(A + (long)r1 * PH, 0, 1);

    v8f acc0, acc1;
    float bv = bias[col];
    #pragma unroll
    for (int j = 0; j < 8; ++j) { acc0[j] = bv; acc1[j] = bv; }

    #pragma unroll 4
    for (int kc = 0; kc < 8; ++kc) {
        int k0 = kc * 32 + hi * 16;
        v16bf a0 = cat8(*(const v8bf*)(A + (long)r0 * PH + k0),
                        *(const v8bf*)(A + (long)r0 * PH + k0 + 8));
        v16bf a1 = cat8(*(const v8bf*)(A + (long)r1 * PH + k0),
                        *(const v8bf*)(A + (long)r1 * PH + k0 + 8));
        v16bf b = *(const v16bf*)(smem + (kc * 32 + lane) * 16);
        acc0 = __builtin_amdgcn_wmma_f32_16x16x32_bf16(false, a0, false, b, (short)0, acc0,
                                                       false, false);
        acc1 = __builtin_amdgcn_wmma_f32_16x16x32_bf16(false, a1, false, b, (short)0, acc1,
                                                       false, false);
    }
    #pragma unroll
    for (int j = 0; j < 8; ++j) {
        int m0 = mPair * 32 + j + hi * 8;
        int m1 = m0 + 16;
        int b0 = m0 >> 9, s0 = m0 & (PS - 1);
        int b1 = m1 >> 9, s1 = m1 & (PS - 1);
        R[((long)b0 * PH + col) * PS + s0] = acc0[j];
        R[((long)b1 * PH + col) * PS + s1] = acc1[j];
    }
}

// Query projection: out[b][h] = sum_k Wt[k][h] * Q[b][k] + bias[h]
// 128 tiles; block = 8 waves (all 8 M-tiles) sharing one N-tile; 16 blocks.
__global__ __launch_bounds__(256) void k_q_gemm(const __bf16* __restrict__ Q,
                                                const __bf16* __restrict__ Wsw,
                                                const float* __restrict__ bias,
                                                float* __restrict__ out) {
    __shared__ __align__(32) __bf16 smem[8 * 512];
    int tid  = threadIdx.x;
    int lane = tid & 31;
    int mTile = tid >> 5;
    int nT    = blockIdx.x;

    #pragma unroll
    for (int p = 0; p < 2; ++p) {
        int c = tid + p * 256;
        int kc = c >> 6;
        int within = c & 63;
        const v8bf* src = (const v8bf*)(Wsw + (((long)kc * 16 + nT) * 512) + within * 8);
        *(v8bf*)(smem + (long)kc * 512 + within * 8) = *src;
    }
    __syncthreads();

    int row = mTile * 16 + (lane & 15);
    int col = nT * 16 + (lane & 15);
    int hi  = lane >> 4;

    v8f acc;
    float bv = bias[col];
    #pragma unroll
    for (int j = 0; j < 8; ++j) acc[j] = bv;

    #pragma unroll
    for (int kc = 0; kc < 8; ++kc) {
        int k0 = kc * 32 + hi * 16;
        v16bf a = cat8(*(const v8bf*)(Q + row * PH + k0),
                       *(const v8bf*)(Q + row * PH + k0 + 8));
        v16bf b = *(const v16bf*)(smem + (kc * 32 + lane) * 16);
        acc = __builtin_amdgcn_wmma_f32_16x16x32_bf16(false, a, false, b, (short)0, acc,
                                                      false, false);
    }
    #pragma unroll
    for (int j = 0; j < 8; ++j) {
        int r = mTile * 16 + j + hi * 8;
        out[r * PH + col] = acc[j];
    }
}

// ---------------- Pointwise LSTM cell ----------------
__global__ __launch_bounds__(256) void k_pointwise(const float* __restrict__ G,
                                                   float* __restrict__ c,
                                                   float* __restrict__ h,
                                                   __bf16* __restrict__ hb,
                                                   __bf16* __restrict__ encOut,
                                                   long encRowStride) {
    int i = blockIdx.x * 256 + threadIdx.x; // 32768 = B*H
    int b = i >> 8, j = i & 255;
    const float* g = G + (long)b * P4H;
    float gi = g[j], gf = g[PH + j], gg = g[2 * PH + j], go = g[3 * PH + j];
    float cn = sigf(gf) * c[i] + sigf(gi) * tanhf(gg);
    float hn = sigf(go) * tanhf(cn);
    c[i] = cn; h[i] = hn;
    __bf16 hv = f2bf(hn);
    hb[i] = hv;
    if (encOut) encOut[(long)b * encRowStride + j] = hv;
}

// ---------------- Attention kernels ----------------
__global__ __launch_bounds__(256) void k_glimpse(const float* __restrict__ qp,
                                                 const float* __restrict__ ref,
                                                 const float* __restrict__ V,
                                                 const unsigned char* __restrict__ mask,
                                                 __bf16* __restrict__ q2b) {
    __shared__ float lg[PS];
    __shared__ float red[8];
    int b = blockIdx.x;
    int tid = threadIdx.x;
    int lane = tid & 31, wv = tid >> 5;
    const float* refb = ref + (long)b * PH * PS;
    const float* qpb  = qp + b * PH;

    for (int s = tid; s < PS; s += 256) {
        float acc = 0.f;
        for (int hh = 0; hh < PH; ++hh)
            acc += V[hh] * tanhf(qpb[hh] + refb[(long)hh * PS + s]);
        lg[s] = mask[b * PS + s] ? -3.0e38f : acc;
    }
    __syncthreads();

    float m = -3.0e38f;
    for (int s = tid; s < PS; s += 256) m = fmaxf(m, lg[s]);
    m = waveRedMax(m);
    if (lane == 0) red[wv] = m;
    __syncthreads();
    m = red[0];
    #pragma unroll
    for (int k = 1; k < 8; ++k) m = fmaxf(m, red[k]);

    float sum = 0.f;
    for (int s = tid; s < PS; s += 256) {
        float p = __expf(lg[s] - m);
        lg[s] = p;
        sum += p;
    }
    sum = waveRedSum(sum);
    __syncthreads();
    if (lane == 0) red[wv] = sum;
    __syncthreads();
    sum = 0.f;
    #pragma unroll
    for (int k = 0; k < 8; ++k) sum += red[k];
    float inv = 1.0f / sum;
    __syncthreads();

    int hh = tid;
    float acc = 0.f;
    const float* r = refb + (long)hh * PS;
    for (int s = 0; s < PS; ++s) acc += r[s] * lg[s];
    q2b[b * PH + hh] = f2bf(acc * inv);
}

__global__ __launch_bounds__(256) void k_ptr_attn(const float* __restrict__ qp,
                                                  const float* __restrict__ ref,
                                                  const float* __restrict__ V,
                                                  unsigned char* __restrict__ mask,
                                                  const __bf16* __restrict__ embBf,
                                                  __bf16* __restrict__ xin,
                                                  float* __restrict__ probsOut,
                                                  int* __restrict__ idxOut,
                                                  int t) {
    __shared__ float lg[PS];
    __shared__ float pe[PS];
    __shared__ float red[8];
    __shared__ float bval[8];
    __shared__ int   bidx[8];
    __shared__ int   sidx;
    int b = blockIdx.x;
    int tid = threadIdx.x;
    int lane = tid & 31, wv = tid >> 5;
    const float* refb = ref + (long)b * PH * PS;
    const float* qpb  = qp + b * PH;

    for (int s = tid; s < PS; s += 256) {
        float acc = 0.f;
        for (int hh = 0; hh < PH; ++hh)
            acc += V[hh] * tanhf(qpb[hh] + refb[(long)hh * PS + s]);
        acc = C_TANH * tanhf(acc);
        lg[s] = mask[b * PS + s] ? -3.0e38f : acc;
    }
    __syncthreads();

    float m = -3.0e38f;
    for (int s = tid; s < PS; s += 256) m = fmaxf(m, lg[s]);
    m = waveRedMax(m);
    if (lane == 0) red[wv] = m;
    __syncthreads();
    m = red[0];
    #pragma unroll
    for (int k = 1; k < 8; ++k) m = fmaxf(m, red[k]);

    float sum = 0.f;
    for (int s = tid; s < PS; s += 256) {
        float p = __expf(lg[s] - m);
        pe[s] = p;
        sum += p;
    }
    sum = waveRedSum(sum);
    __syncthreads();
    if (lane == 0) red[wv] = sum;
    __syncthreads();
    sum = 0.f;
    #pragma unroll
    for (int k = 0; k < 8; ++k) sum += red[k];
    float inv = 1.0f / sum;

    float* po = probsOut + (long)t * PB * PS + (long)b * PS;
    for (int s = tid; s < PS; s += 256) po[s] = pe[s] * inv;

    float bestV = -3.0e38f;
    int   bestI = 0;
    for (int s = tid; s < PS; s += 256) {
        unsigned hsh = hash3((unsigned)t, (unsigned)b, (unsigned)s);
        float u = ((hsh >> 8) + 1.0f) * (1.0f / 16777218.0f);
        float g = (lg[s] <= -1.0e38f) ? -3.0e38f : lg[s] - logf(-logf(u));
        if (g > bestV) { bestV = g; bestI = s; }
    }
    #pragma unroll
    for (int o = 16; o > 0; o >>= 1) {
        float ov = __shfl_xor(bestV, o, 32);
        int   oi = __shfl_xor(bestI, o, 32);
        if (ov > bestV || (ov == bestV && oi < bestI)) { bestV = ov; bestI = oi; }
    }
    if (lane == 0) { bval[wv] = bestV; bidx[wv] = bestI; }
    __syncthreads();
    if (tid == 0) {
        float bv = bval[0]; int bi = bidx[0];
        #pragma unroll
        for (int k = 1; k < 8; ++k)
            if (bval[k] > bv || (bval[k] == bv && bidx[k] < bi)) { bv = bval[k]; bi = bidx[k]; }
        sidx = bi;
        idxOut[(long)t * PB + b] = bi;
        mask[b * PS + bi] = 1;  // becomes effective at step t+1 (matches reference)
    }
    __syncthreads();
    int si = sidx;
    if (tid < PE)
        xin[b * PE + tid] = embBf[(long)b * PS * PE + (long)si * PE + tid];
}

// ---------------- Host orchestration ----------------
extern "C" void kernel_launch(void* const* d_in, const int* in_sizes, int n_in,
                              void* d_out, int out_size, void* d_ws, size_t ws_size,
                              hipStream_t stream) {
    (void)in_sizes; (void)n_in; (void)out_size; (void)ws_size;

    const float* inputs   = (const float*)d_in[0];
    const float* emb      = (const float*)d_in[1];
    const float* enc_Wih  = (const float*)d_in[2];
    const float* enc_Whh  = (const float*)d_in[3];
    const float* enc_bih  = (const float*)d_in[4];
    const float* enc_bhh  = (const float*)d_in[5];
    const float* dec_Wih  = (const float*)d_in[6];
    const float* dec_Whh  = (const float*)d_in[7];
    const float* dec_bih  = (const float*)d_in[8];
    const float* dec_bhh  = (const float*)d_in[9];
    const float* ptr_Wq   = (const float*)d_in[10];
    const float* ptr_bq   = (const float*)d_in[11];
    const float* ptr_Wref = (const float*)d_in[12];
    const float* ptr_bref = (const float*)d_in[13];
    const float* ptr_V    = (const float*)d_in[14];
    const float* gl_Wq    = (const float*)d_in[15];
    const float* gl_bq    = (const float*)d_in[16];
    const float* gl_Wref  = (const float*)d_in[17];
    const float* gl_bref  = (const float*)d_in[18];
    const float* gl_V     = (const float*)d_in[19];
    const float* dec_in0  = (const float*)d_in[20];

    char* ws = (char*)d_ws;
    size_t off = 0;
    auto carve = [&](size_t bytes) -> char* {
        char* p = ws + off;
        off = (off + bytes + 511) & ~(size_t)511;
        return p;
    };
    __bf16* embBf   = (__bf16*)carve((size_t)PB * PS * PE * 2);
    __bf16* encOut  = (__bf16*)carve((size_t)PB * PS * PH * 2);
    float*  ptrRef  = (float*) carve((size_t)PB * PH * PS * 4);    // L2-resident
    float*  glRef   = (float*) carve((size_t)PB * PH * PS * 4);    // L2-resident
    __bf16* WcatE   = (__bf16*)carve((size_t)PKCAT * P4H * 2);
    __bf16* WcatD   = (__bf16*)carve((size_t)PKCAT * P4H * 2);
    float*  bsumE   = (float*) carve((size_t)P4H * 4);
    float*  bsumD   = (float*) carve((size_t)P4H * 4);
    __bf16* glWqT   = (__bf16*)carve((size_t)PH * PH * 2);
    __bf16* ptrWqT  = (__bf16*)carve((size_t)PH * PH * 2);
    __bf16* glWrT   = (__bf16*)carve((size_t)PH * PH * 2);
    __bf16* ptrWrT  = (__bf16*)carve((size_t)PH * PH * 2);
    float*  hBuf    = (float*) carve((size_t)PB * PH * 4);
    float*  cBuf    = (float*) carve((size_t)PB * PH * 4);
    __bf16* hbBuf   = (__bf16*)carve((size_t)PB * PH * 2);
    float*  Gbuf    = (float*) carve((size_t)PB * P4H * 4);
    float*  qpBuf   = (float*) carve((size_t)PB * PH * 4);
    float*  qp2Buf  = (float*) carve((size_t)PB * PH * 4);
    __bf16* q2bBuf  = (__bf16*)carve((size_t)PB * PH * 2);
    __bf16* xinBuf  = (__bf16*)carve((size_t)PB * PE * 2);
    unsigned char* maskBuf = (unsigned char*)carve((size_t)PB * PS);

    float* probsOut = (float*)d_out;                                 // [S,B,S] fp32
    int*   idxOut   = (int*)((float*)d_out + (long)PS * PB * PS);    // [S,B] int32

    // ---- preparation ----
    k_embed<<<(PB * PS * PE) / 256, 256, 0, stream>>>(inputs, emb, embBf);
    k_prep_cat<<<(PKCAT * P4H + 255) / 256, 256, 0, stream>>>(enc_Wih, enc_Whh, enc_bih,
                                                              enc_bhh, WcatE, bsumE);
    k_prep_cat<<<(PKCAT * P4H + 255) / 256, 256, 0, stream>>>(dec_Wih, dec_Whh, dec_bih,
                                                              dec_bhh, WcatD, bsumD);
    k_prep_sq<<<(PH * PH) / 256, 256, 0, stream>>>(gl_Wq, glWqT);
    k_prep_sq<<<(PH * PH) / 256, 256, 0, stream>>>(ptr_Wq, ptrWqT);
    k_prep_sq<<<(PH * PH) / 256, 256, 0, stream>>>(gl_Wref, glWrT);
    k_prep_sq<<<(PH * PH) / 256, 256, 0, stream>>>(ptr_Wref, ptrWrT);
    k_init<<<(PB * PS) / 256, 256, 0, stream>>>(hBuf, cBuf, hbBuf, maskBuf);
    k_setxin<<<(PB * PE) / 256, 256, 0, stream>>>(dec_in0, xinBuf);

    // ---- encoder LSTM (WMMA gate GEMM per step) ----
    for (int t = 0; t < PS; ++t) {
        k_gates<<<32, 256, 0, stream>>>(embBf + (long)t * PE, (long)PS * PE, hbBuf,
                                        WcatE, bsumE, Gbuf);
        k_pointwise<<<128, 256, 0, stream>>>(Gbuf, cBuf, hBuf, hbBuf,
                                             encOut + (long)t * PH, (long)PS * PH);
    }

    // ---- loop-invariant ref projections (large parallel WMMA GEMMs) ----
    k_ref_gemm<<<4096, 256, 0, stream>>>(encOut, ptrWrT, ptr_bref, ptrRef);
    k_ref_gemm<<<4096, 256, 0, stream>>>(encOut, glWrT, gl_bref, glRef);

    // ---- decoder ----
    for (int t = 0; t < PS; ++t) {
        k_gates<<<32, 256, 0, stream>>>(xinBuf, (long)PE, hbBuf, WcatD, bsumD, Gbuf);
        k_pointwise<<<128, 256, 0, stream>>>(Gbuf, cBuf, hBuf, hbBuf,
                                             (__bf16*)nullptr, 0L);
        k_q_gemm<<<16, 256, 0, stream>>>(hbBuf, glWqT, gl_bq, qpBuf);
        k_glimpse<<<PB, 256, 0, stream>>>(qpBuf, glRef, gl_V, maskBuf, q2bBuf);
        k_q_gemm<<<16, 256, 0, stream>>>(q2bBuf, ptrWqT, ptr_bq, qp2Buf);
        k_ptr_attn<<<PB, 256, 0, stream>>>(qp2Buf, ptrRef, ptr_V, maskBuf, embBf, xinBuf,
                                           probsOut, idxOut, t);
    }
}
